// NeuronGatedDeltaNet_62070867361982
// MI455X (gfx1250) — compile-verified, matrix operand order checked by reading, not run on hardware
//
#include <hip/hip_runtime.h>

// ---------------------------------------------------------------------------
// Gated DeltaNet forward for MI455X (gfx1250, wave32, WMMA).
//  * All GEMM operands pre-converted to bf16 once (halves weight DRAM traffic,
//    removes cvt from GEMM hot loop).
//  * GEMM: v_wmma_f32_16x16x32_bf16, BK=64, double-buffered LDS with the
//    gfx1250 async global->LDS DMA pipelined against WMMA via
//    s_wait_asynccnt(8) (async loads retire in order).
//  * Delta-rule scan: state in VGPRs (64 f32/thread), LDS broadcasts/reduces.
// ---------------------------------------------------------------------------

#define Bn       2
#define Sn       2048
#define HIDN     2048
#define HVh      32
#define HKh      16
#define DKd      128
#define DVd      128
#define KEY_DIM  2048
#define VAL_DIM  4096
#define CONV_DIM 8192
#define KSz      4

typedef __attribute__((ext_vector_type(16))) __bf16 v16bf;
typedef __attribute__((ext_vector_type(8)))  float  v8f;
typedef __attribute__((ext_vector_type(4)))  unsigned int v4u;
typedef __attribute__((ext_vector_type(4)))  int v4i;

struct FragBits { v4u lo, hi; };

#if __has_builtin(__builtin_amdgcn_global_load_async_to_lds_b128) && \
    __has_builtin(__builtin_amdgcn_s_wait_asynccnt)
#define USE_ASYNC_COPY 1
typedef __attribute__((address_space(1))) v4i* as1_v4i_ptr;
typedef __attribute__((address_space(3))) v4i* as3_v4i_ptr;
#else
#define USE_ASYNC_COPY 0
#endif

__device__ __forceinline__ unsigned short f32_to_bf16(float f) {
  unsigned int u = __float_as_uint(f);
  u += 0x7FFFu + ((u >> 16) & 1u);           // round-to-nearest-even
  return (unsigned short)(u >> 16);
}

// 16-byte global -> LDS copy (async LDS-DMA on gfx1250 if available).
__device__ __forceinline__ void copy16_g2l(const unsigned short* g, unsigned short* l) {
#if USE_ASYNC_COPY
  __builtin_amdgcn_global_load_async_to_lds_b128(
      (as1_v4i_ptr)g, (as3_v4i_ptr)l, 0, 0);
#else
  *reinterpret_cast<v4u*>(l) = *reinterpret_cast<const v4u*>(g);
#endif
}

// Wait until all async copies done (end of pipeline).
__device__ __forceinline__ void copy_wait_all() {
#if USE_ASYNC_COPY
  __builtin_amdgcn_s_wait_asynccnt(0);
#endif
}
// Wait until only the 8 newest async copies remain in flight (steady state:
// this thread's 8 ops for the *current* tile are complete, next tile's 8
// keep streaming behind the WMMAs).
__device__ __forceinline__ void copy_wait_tile() {
#if USE_ASYNC_COPY
  __builtin_amdgcn_s_wait_asynccnt(8);
#endif
}

__device__ __forceinline__ v16bf ld_frag(const unsigned short* lds, int base) {
  FragBits fb;
  fb.lo = *reinterpret_cast<const v4u*>(lds + base);        // K[half*8 .. +7]
  fb.hi = *reinterpret_cast<const v4u*>(lds + base + 16);   // K[16+half*8 .. +7]
  return __builtin_bit_cast(v16bf, fb);
}

// ---------------------------------------------------------------------------
// fp32 -> packed bf16 (8 elems/thread); zero-fills pad tail (n_src..n_dst).
// ---------------------------------------------------------------------------
__global__ __launch_bounds__(256) void to_bf16_kernel(
    const float* __restrict__ src, unsigned short* __restrict__ dst,
    long n_src, long n_dst) {
  const long i8 = ((long)blockIdx.x * 256 + threadIdx.x) * 8;
  if (i8 >= n_dst) return;
  float v[8];
  if (i8 + 8 <= n_src) {
    const float4* p = reinterpret_cast<const float4*>(src + i8);
    const float4 f0 = p[0], f1 = p[1];
    v[0] = f0.x; v[1] = f0.y; v[2] = f0.z; v[3] = f0.w;
    v[4] = f1.x; v[5] = f1.y; v[6] = f1.z; v[7] = f1.w;
  } else {
#pragma unroll
    for (int j = 0; j < 8; ++j) v[j] = 0.f;
  }
  uint4 o;
  o.x = (unsigned)f32_to_bf16(v[0]) | ((unsigned)f32_to_bf16(v[1]) << 16);
  o.y = (unsigned)f32_to_bf16(v[2]) | ((unsigned)f32_to_bf16(v[3]) << 16);
  o.z = (unsigned)f32_to_bf16(v[4]) | ((unsigned)f32_to_bf16(v[5]) << 16);
  o.w = (unsigned)f32_to_bf16(v[6]) | ((unsigned)f32_to_bf16(v[7]) << 16);
  *reinterpret_cast<uint4*>(dst + i8) = o;
}

// ---------------------------------------------------------------------------
// C[M,N] = A[M,K] @ W[N,K]^T  (A,W bf16; C fp32). M,N multiples of 128,
// K multiple of 64. Block tile 128x128, 8 waves (2x4), wave tile 64x32,
// BK=64 -> 16 WMMAs per stage; double-buffered LDS, async DMA pipelined.
// ---------------------------------------------------------------------------
__global__ __launch_bounds__(256) void gemm_bf16_wmma(
    const unsigned short* __restrict__ A, const unsigned short* __restrict__ W,
    float* __restrict__ C, int M, int N, int K) {
  __shared__ __align__(16) unsigned short As[2][128 * 64];
  __shared__ __align__(16) unsigned short Bs[2][128 * 64];

  const int tid  = threadIdx.x;
  const int lane = tid & 31;
  const int wave = tid >> 5;
  const int wm   = (wave >> 2) * 64;   // wave M offset in block tile
  const int wn   = (wave & 3) * 32;    // wave N offset
  const int m0   = blockIdx.y * 128;
  const int n0   = blockIdx.x * 128;
  const int half = lane >> 4;
  const int r16  = lane & 15;

  v8f acc[4][2];
#pragma unroll
  for (int i = 0; i < 4; ++i)
#pragma unroll
    for (int j = 0; j < 2; ++j) acc[i][j] = v8f{};

  // Stage a 128x64 bf16 tile pair into LDS buffer `buf` (8 b128 ops/thread).
  auto stage = [&](int buf, int kt) {
    const unsigned short* Ag = A + (size_t)m0 * K + kt * 64;
    const unsigned short* Wg = W + (size_t)n0 * K + kt * 64;
#pragma unroll
    for (int it = 0; it < 4; ++it) {
      const int ci = tid + it * 256;          // 0..1023
      const int r  = ci >> 3;
      const int cc = (ci & 7) * 8;
      copy16_g2l(Ag + (size_t)r * K + cc, &As[buf][r * 64 + cc]);
      copy16_g2l(Wg + (size_t)r * K + cc, &Bs[buf][r * 64 + cc]);
    }
  };

  const int nk = K >> 6;
  stage(0, 0);                               // prologue

  for (int kt = 0; kt < nk; ++kt) {
    const int cur = kt & 1;
    if (kt + 1 < nk) {
      stage(cur ^ 1, kt + 1);                // overlap next tile's DMA
      copy_wait_tile();                      // current tile's 8 ops complete
    } else {
      copy_wait_all();
    }
    __syncthreads();

#pragma unroll
    for (int ks = 0; ks < 2; ++ks) {
      v16bf af[4], bfr[2];
#pragma unroll
      for (int i = 0; i < 4; ++i)
        af[i] = ld_frag(&As[cur][0], (wm + i * 16 + r16) * 64 + ks * 32 + half * 8);
#pragma unroll
      for (int j = 0; j < 2; ++j)
        bfr[j] = ld_frag(&Bs[cur][0], (wn + j * 16 + r16) * 64 + ks * 32 + half * 8);
#pragma unroll
      for (int i = 0; i < 4; ++i)
#pragma unroll
        for (int j = 0; j < 2; ++j)
          acc[i][j] = __builtin_amdgcn_wmma_f32_16x16x32_bf16(
              false, af[i], false, bfr[j], (short)0, acc[i][j], false, false);
    }
    __syncthreads();                         // all waves done with buf `cur`
  }

  // C/D layout: lanes 0-15: M=r, N=lane; lanes 16-31: M=8+r, N=lane-16.
#pragma unroll
  for (int i = 0; i < 4; ++i)
#pragma unroll
    for (int j = 0; j < 2; ++j)
#pragma unroll
      for (int r = 0; r < 8; ++r) {
        const int row = m0 + wm + i * 16 + half * 8 + r;
        const int col = n0 + wn + j * 16 + r16;
        C[(size_t)row * N + col] = acc[i][j][r];
      }
}

// ---------------------------------------------------------------------------
// Causal depthwise conv1d (K=4, left pad 3) + SiLU. One block per (b,s) row.
// ---------------------------------------------------------------------------
__global__ __launch_bounds__(256) void conv_silu_kernel(
    const float* __restrict__ qkv, const float* __restrict__ cw,
    float* __restrict__ mixed) {
  const int row = blockIdx.x;            // b*Sn + s
  const int s   = row & (Sn - 1);
  for (int c = threadIdx.x; c < CONV_DIM; c += 256) {
    float acc = 0.f;
#pragma unroll
    for (int k = 0; k < KSz; ++k) {
      const int tt = s + k - (KSz - 1);
      if (tt >= 0)
        acc += qkv[(size_t)(row + k - (KSz - 1)) * CONV_DIM + c] * cw[c * KSz + k];
    }
    mixed[(size_t)row * CONV_DIM + c] = acc / (1.f + __expf(-acc));
  }
}

// ---------------------------------------------------------------------------
// In-place l2norm of q,k head vectors; q additionally scaled by DK^-0.5.
// ---------------------------------------------------------------------------
__global__ __launch_bounds__(128) void qk_l2norm_kernel(float* __restrict__ mixed) {
  const int idx = blockIdx.x;            // (b*Sn+s)*HK + hk
  const int row = idx >> 4;
  const int hk  = idx & 15;
  const int t   = threadIdx.x;
  float* qp = mixed + (size_t)row * CONV_DIM + hk * DKd;
  float* kp = qp + KEY_DIM;
  __shared__ float red[128];

  const float qv = qp[t];
  red[t] = qv * qv; __syncthreads();
  for (int off = 64; off; off >>= 1) { if (t < off) red[t] += red[t + off]; __syncthreads(); }
  const float qn = red[0]; __syncthreads();

  const float kv = kp[t];
  red[t] = kv * kv; __syncthreads();
  for (int off = 64; off; off >>= 1) { if (t < off) red[t] += red[t + off]; __syncthreads(); }
  const float kn = red[0];

  qp[t] = qv / fmaxf(sqrtf(qn), 1e-6f) * 0.08838834764831845f;  // DK^-0.5
  kp[t] = kv / fmaxf(sqrtf(kn), 1e-6f);
}

// ---------------------------------------------------------------------------
// g = -exp(A_log[h]) * softplus(a + dt_bias[h]);  beta = sigmoid(b)
// a/b come from 128-col padded projection outputs.
// ---------------------------------------------------------------------------
__global__ __launch_bounds__(256) void gates_kernel(
    const float* __restrict__ a, const float* __restrict__ b,
    const float* __restrict__ A_log, const float* __restrict__ dt_bias,
    float* __restrict__ g, float* __restrict__ beta) {
  const int i = blockIdx.x * 256 + threadIdx.x;
  if (i >= Bn * Sn * HVh) return;
  const int row = i >> 5;
  const int h   = i & (HVh - 1);
  const float x  = a[row * 128 + h] + dt_bias[h];
  const float sp = (x > 20.f) ? x : log1pf(__expf(x));
  g[i]    = -__expf(A_log[h]) * sp;
  const float bv = b[row * 128 + h];
  beta[i] = 1.f / (1.f + __expf(-bv));
}

// ---------------------------------------------------------------------------
// Sequential delta-rule recurrence. One block per (b,h); 256 threads; state
// S[128,128] lives in VGPRs: thread (half,v) owns S[half*64..+63, v].
// ---------------------------------------------------------------------------
__global__ __launch_bounds__(256) void delta_recurrence_kernel(
    const float* __restrict__ mixed, const float* __restrict__ g,
    const float* __restrict__ beta, float* __restrict__ o) {
  const int bh   = blockIdx.x;           // b*HV + h
  const int b    = bh >> 5;
  const int h    = bh & 31;
  const int hk   = h >> 1;               // repeat-interleave: rep = HV/HK = 2
  const int t    = threadIdx.x;
  const int v    = t & 127;
  const int half = t >> 7;
  const int kb   = half * 64;

  __shared__ float qsh[128], ksh[128], vsh[128], dsh[128], red[256];
  float st[64];
#pragma unroll
  for (int i = 0; i < 64; ++i) st[i] = 0.f;

  for (int s = 0; s < Sn; ++s) {
    const float* mrow = mixed + ((size_t)b * Sn + s) * CONV_DIM;
    if (t < 128) {
      qsh[t] = mrow[hk * DKd + t];
      ksh[t] = mrow[KEY_DIM + hk * DKd + t];
    } else {
      vsh[t & 127] = mrow[2 * KEY_DIM + h * DVd + (t & 127)];
    }
    __syncthreads();

    const size_t gi  = ((size_t)b * Sn + s) * HVh + h;
    const float  dec = __expf(g[gi]);
    const float  bt  = beta[gi];

    // decay + kv = k^T S
    float kvp = 0.f;
#pragma unroll
    for (int kk = 0; kk < 64; ++kk) {
      const float sv = st[kk] * dec;
      st[kk] = sv;
      kvp += ksh[kb + kk] * sv;
    }
    red[t] = kvp;
    __syncthreads();
    if (t < 128) dsh[t] = (vsh[t] - (red[t] + red[t + 128])) * bt;
    __syncthreads();

    // rank-1 write + out = q^T S
    const float dv = dsh[v];
    float qop = 0.f;
#pragma unroll
    for (int kk = 0; kk < 64; ++kk) {
      const float sv = st[kk] + ksh[kb + kk] * dv;
      st[kk] = sv;
      qop += qsh[kb + kk] * sv;
    }
    red[t] = qop;
    __syncthreads();
    if (t < 128) o[gi * DVd + t] = red[t] + red[t + 128];
    __syncthreads();
  }
}

// ---------------------------------------------------------------------------
// Gated per-head RMSNorm + SiLU(z) gate; emits bf16 for the final WMMA GEMM.
// ---------------------------------------------------------------------------
__global__ __launch_bounds__(128) void rmsnorm_gate_kernel(
    const float* __restrict__ o, const float* __restrict__ z,
    const float* __restrict__ norm_w, unsigned short* __restrict__ of_bf) {
  const int idx = blockIdx.x;            // (b*Sn+s)*HV + h
  const int row = idx >> 5;
  const int h   = idx & 31;
  const int t   = threadIdx.x;
  const float ov = o[(size_t)idx * DVd + t];
  __shared__ float red[128];
  red[t] = ov * ov; __syncthreads();
  for (int off = 64; off; off >>= 1) { if (t < off) red[t] += red[t + off]; __syncthreads(); }
  const float ms = red[0] * (1.f / 128.f);
  float val = ov * rsqrtf(ms + 1e-6f) * norm_w[t];
  const float zv = z[(size_t)row * VAL_DIM + h * DVd + t];
  val *= zv / (1.f + __expf(-zv));
  of_bf[(size_t)row * VAL_DIM + h * DVd + t] = f32_to_bf16(val);
}

// ---------------------------------------------------------------------------
extern "C" void kernel_launch(void* const* d_in, const int* in_sizes, int n_in,
                              void* d_out, int out_size, void* d_ws, size_t ws_size,
                              hipStream_t stream) {
  const float* hs      = (const float*)d_in[0];
  const float* W_qkv   = (const float*)d_in[1];
  const float* W_z     = (const float*)d_in[2];
  const float* W_a     = (const float*)d_in[3];
  const float* W_b     = (const float*)d_in[4];
  const float* conv_w  = (const float*)d_in[5];
  const float* A_log   = (const float*)d_in[6];
  const float* dt_bias = (const float*)d_in[7];
  const float* norm_w  = (const float*)d_in[8];
  const float* W_out   = (const float*)d_in[9];
  float* out = (float*)d_out;

  const int M = Bn * Sn;                         // 4096
  char* p = (char*)d_ws;
  const size_t QKV_B = (size_t)M * CONV_DIM * 4; // 134 MB

  float* qkv    = (float*)p;  p += QKV_B;
  float* mixed  = (float*)p;  p += QKV_B;
  float* z      = (float*)p;  p += (size_t)M * VAL_DIM * 4;
  float* a_p    = (float*)p;  p += (size_t)M * 128 * 4;     // padded 128 cols
  float* b_p    = (float*)p;  p += (size_t)M * 128 * 4;
  float* g_p    = (float*)p;  p += (size_t)M * HVh * 4;
  float* beta_p = (float*)p;  p += (size_t)M * HVh * 4;
  unsigned short* hs_bf   = (unsigned short*)p;  p += (size_t)M * HIDN * 2;
  unsigned short* wqkv_bf = (unsigned short*)p;  p += (size_t)CONV_DIM * HIDN * 2;
  unsigned short* wz_bf   = (unsigned short*)p;  p += (size_t)VAL_DIM * HIDN * 2;
  unsigned short* wa_bf   = (unsigned short*)p;  p += (size_t)128 * HIDN * 2;  // padded rows
  unsigned short* wb_bf   = (unsigned short*)p;  p += (size_t)128 * HIDN * 2;
  unsigned short* wout_bf = (unsigned short*)p;  p += (size_t)HIDN * VAL_DIM * 2;
  unsigned short* of_bf   = (unsigned short*)p;  p += (size_t)M * VAL_DIM * 2;
  float* o_p = (float*)d_ws;                     // reuse qkv region (dead after conv)

  dim3 blk(256);
  auto cvt_grid = [](long nd) { return (unsigned)((nd / 8 + 255) / 256); };

  // One-time fp32 -> bf16 conversions (pad a/b weights to 128 rows).
  to_bf16_kernel<<<cvt_grid((long)M * HIDN), blk, 0, stream>>>(hs, hs_bf, (long)M * HIDN, (long)M * HIDN);
  to_bf16_kernel<<<cvt_grid((long)CONV_DIM * HIDN), blk, 0, stream>>>(W_qkv, wqkv_bf, (long)CONV_DIM * HIDN, (long)CONV_DIM * HIDN);
  to_bf16_kernel<<<cvt_grid((long)VAL_DIM * HIDN), blk, 0, stream>>>(W_z, wz_bf, (long)VAL_DIM * HIDN, (long)VAL_DIM * HIDN);
  to_bf16_kernel<<<cvt_grid((long)128 * HIDN), blk, 0, stream>>>(W_a, wa_bf, (long)HVh * HIDN, (long)128 * HIDN);
  to_bf16_kernel<<<cvt_grid((long)128 * HIDN), blk, 0, stream>>>(W_b, wb_bf, (long)HVh * HIDN, (long)128 * HIDN);
  to_bf16_kernel<<<cvt_grid((long)HIDN * VAL_DIM), blk, 0, stream>>>(W_out, wout_bf, (long)HIDN * VAL_DIM, (long)HIDN * VAL_DIM);

  // Input projections (WMMA).
  gemm_bf16_wmma<<<dim3(CONV_DIM / 128, M / 128), blk, 0, stream>>>(hs_bf, wqkv_bf, qkv, M, CONV_DIM, HIDN);
  gemm_bf16_wmma<<<dim3(VAL_DIM / 128,  M / 128), blk, 0, stream>>>(hs_bf, wz_bf,   z,   M, VAL_DIM,  HIDN);
  gemm_bf16_wmma<<<dim3(1,              M / 128), blk, 0, stream>>>(hs_bf, wa_bf,   a_p, M, 128,      HIDN);
  gemm_bf16_wmma<<<dim3(1,              M / 128), blk, 0, stream>>>(hs_bf, wb_bf,   b_p, M, 128,      HIDN);

  conv_silu_kernel<<<M, 256, 0, stream>>>(qkv, conv_w, mixed);
  qk_l2norm_kernel<<<M * HKh, 128, 0, stream>>>(mixed);
  gates_kernel<<<(M * HVh + 255) / 256, 256, 0, stream>>>(a_p, b_p, A_log, dt_bias, g_p, beta_p);
  delta_recurrence_kernel<<<Bn * HVh, 256, 0, stream>>>(mixed, g_p, beta_p, o_p);
  rmsnorm_gate_kernel<<<M * HVh, 128, 0, stream>>>(o_p, z, norm_w, of_bf);

  // Output projection (WMMA) straight into d_out.
  gemm_bf16_wmma<<<dim3(HIDN / 128, M / 128), blk, 0, stream>>>(of_bf, wout_bf, out, M, HIDN, VAL_DIM);
}
